// OrientedRPN_23562190586563
// MI455X (gfx1250) — matrix operand order repaired and checked by reading, not compiled, not run on hardware
//
#include <hip/hip_runtime.h>

// ---------------------------------------------------------------------------
// Types for CDNA5 WMMA
// ---------------------------------------------------------------------------
typedef __bf16 v16bf __attribute__((ext_vector_type(16)));
typedef float  v8f   __attribute__((ext_vector_type(8)));

typedef union {
    v16bf v;
    unsigned int u[8];
} FragAB;

// ---------------------------------------------------------------------------
// Small helpers
// ---------------------------------------------------------------------------
static __device__ __forceinline__ unsigned short f2bf(float f) {
    unsigned int u = __float_as_uint(f);
    unsigned int r = u + 0x7FFFu + ((u >> 16) & 1u);   // round-to-nearest-even
    return (unsigned short)(r >> 16);
}
static __device__ __forceinline__ float bf2f(unsigned short h) {
    return __uint_as_float(((unsigned int)h) << 16);
}
static __device__ __forceinline__ unsigned int pack2(unsigned short a, unsigned short b) {
    return (unsigned int)a | ((unsigned int)b << 16);
}
static __device__ __forceinline__ unsigned int score_key(float s) {
    unsigned int u = __float_as_uint(s);
    return (u & 0x80000000u) ? ~u : (u | 0x80000000u);  // monotonic, larger = better
}

#define AANCH 3
#define KDIM  2304        // 9 taps * 256 in-channels, TAP-MAJOR: k = tap*256 + c
#define TOTROWS 8768      // 2000*4 + 768
#define LDSPITCH 20       // dwords per LDS row (pad: conflict avoidance, 16B-aligned)

// ---------------------------------------------------------------------------
// Kernel 1: repack conv weights fp32 (5,256,256,3,3) -> bf16 [lvl][n][k]
// TAP-MAJOR: k = tap*256 + c ; n-major so B tiles are dword-loadable.
// ---------------------------------------------------------------------------
__global__ void orpn_repack_w(const float* __restrict__ w_conv,
                              unsigned short* __restrict__ wB) {
    const int gid = blockIdx.x * 256 + threadIdx.x;
    const int total = 5 * 256 * KDIM;
    if (gid >= total) return;
    const int lvl = gid / (256 * KDIM);
    const int r0  = gid - lvl * 256 * KDIM;
    const int n   = r0 / KDIM;
    const int k   = r0 - n * KDIM;
    const int r   = k >> 8;        // tap 0..8
    const int c   = k & 255;       // channel
    const float v = w_conv[(((size_t)lvl * 256 + n) * 256 + c) * 9 + r];
    wB[gid] = f2bf(v);
}

// ---------------------------------------------------------------------------
// Kernel 2: 3x3 conv (SAME) + bias + ReLU via implicit GEMM with WMMA bf16.
// M = B*H*W (tile 128), N = 256 (tile 128), K = 2304 tap-major.
// 256 threads = 8 waves (4 M-subtiles x 2 N-subtiles); wave = 32x64 = 2x4 wmma.
// Tap-major K: each 32-wide K step has ONE uniform tap -> per-thread spatial
// decomposition hoisted out of the K loop; uniform dy/dx on SALU.
// ---------------------------------------------------------------------------
__global__ __launch_bounds__(256) void orpn_conv3x3_wmma(
    const float* __restrict__ x, const unsigned short* __restrict__ wB,
    const float* __restrict__ bias, unsigned short* __restrict__ t,
    int H, int W)
{
    const int HWl   = H * W;
    const int mtile = blockIdx.x * 128;
    const int ntile = blockIdx.y * 128;

    __shared__ unsigned int ldsA[128 * LDSPITCH];  // 128 rows x 16 k-pairs (+pad)
    __shared__ unsigned int ldsB[128 * LDSPITCH];  // 128 cols x 16 k-pairs (+pad)

    const int tid  = threadIdx.x;
    const int lane = tid & 31;
    const int wv   = tid >> 5;
    const int wm   = wv & 3;     // 4 M sub-tiles of 32
    const int wn   = wv >> 2;    // 2 N sub-tiles of 64
    const int lr   = lane & 15;
    const int hk   = lane >> 4;

    // ---- hoisted per-thread A-row decomposition (row fixed for whole K loop)
    const int arow_ld = tid & 127;          // A row this thread loads
    const int q       = tid >> 7;           // base k-pair parity (0/1)
    {
    }
    const int m_ld  = mtile + arow_ld;
    const int b_ld  = m_ld / HWl;
    const int hw_ld = m_ld - b_ld * HWl;
    const int oh    = hw_ld / W;
    const int ow    = hw_ld - oh * W;
    const size_t xbase = (size_t)b_ld * 256 * HWl + (size_t)oh * W + ow;

    // B-load mapping (fixed): kp fixed, 8 n-rows
    const int bkp = tid & 15;
    const int bn0 = tid >> 4;

    v8f acc[2][4] = {};

    for (int kt = 0; kt < KDIM; kt += 32) {
        // uniform tap for this K step (SALU)
        const int r  = kt >> 8;
        const int dy = r / 3 - 1;
        const int dx = r - (r / 3) * 3 - 1;
        const int c0 = kt & 255;

        const int iy = oh + dy;
        const int ix = ow + dx;
        const bool valid = ((unsigned)iy < (unsigned)H) && ((unsigned)ix < (unsigned)W);
        const size_t sp = xbase + (size_t)dy * W + dx;

        // ---- A tile: thread loads 8 k-pairs of its fixed row (coalesced) ----
#pragma unroll
        for (int i = 0; i < 8; ++i) {
            const int kp = q + 2 * i;
            const int c  = c0 + kp * 2;
            float v0 = 0.f, v1 = 0.f;
            if (valid) {
                v0 = x[sp + (size_t)c * HWl];
                v1 = x[sp + (size_t)(c + 1) * HWl];
            }
            ldsA[arow_ld * LDSPITCH + kp] = pack2(f2bf(v0), f2bf(v1));
        }
        // ---- B tile: 128 n x 16 k-pairs (contiguous k per n) ----
#pragma unroll
        for (int i = 0; i < 8; ++i) {
            const int n = bn0 + 16 * i;
            const unsigned int* src =
                (const unsigned int*)(wB + (size_t)(ntile + n) * KDIM + kt);
            ldsB[n * LDSPITCH + bkp] = src[bkp];
        }
        // prefetch next weight chunk (global_prefetch_b8)
        if (kt + 32 < KDIM)
            __builtin_prefetch(wB + (size_t)(ntile + (tid & 127)) * KDIM + kt + 32, 0, 1);
        __syncthreads();

        // ---- fragment assembly per CDNA5 16-bit A/B VGPR layouts ----
        FragAB aF[2], bF[4];
#pragma unroll
        for (int mi = 0; mi < 2; ++mi) {
            const int ar = wm * 32 + mi * 16 + lr;
#pragma unroll
            for (int j = 0; j < 8; ++j) {
                const int kpair = (j & 3) + (hk << 2) + ((j >> 2) << 3);
                aF[mi].u[j] = ldsA[ar * LDSPITCH + kpair];
            }
        }
#pragma unroll
        for (int ni = 0; ni < 4; ++ni) {
            const int bn = wn * 64 + ni * 16 + lr;
#pragma unroll
            for (int j = 0; j < 8; ++j)
                bF[ni].u[j] = ldsB[bn * LDSPITCH + (j & 3) + (hk << 2) + ((j >> 2) << 3)];
        }

#pragma unroll
        for (int mi = 0; mi < 2; ++mi)
#pragma unroll
            for (int ni = 0; ni < 4; ++ni)
                acc[mi][ni] = __builtin_amdgcn_wmma_f32_16x16x32_bf16(
                    false, aF[mi].v, false, bF[ni].v,
                    (short)0, acc[mi][ni], false, false);

        __syncthreads();
    }

    // ---- epilogue: bias + ReLU, store bf16 (C/D layout: M=r+8*hk, N=lr) ----
#pragma unroll
    for (int mi = 0; mi < 2; ++mi)
#pragma unroll
        for (int ni = 0; ni < 4; ++ni) {
            const int nc = ntile + wn * 64 + ni * 16 + lr;
            const float bv = bias[nc];
#pragma unroll
            for (int r = 0; r < 8; ++r) {
                const int mr = mtile + wm * 32 + mi * 16 + r + (hk << 3);
                float v = acc[mi][ni][r] + bv;
                v = v > 0.f ? v : 0.f;
                t[(size_t)mr * 256 + nc] = f2bf(v);
            }
        }
}

// ---------------------------------------------------------------------------
// Kernel 3: fused 1x1 heads (reg 18ch + obj 3ch) + anchor box decode.
// ---------------------------------------------------------------------------
__global__ __launch_bounds__(128) void orpn_head_decode(
    const unsigned short* __restrict__ t,
    const float* __restrict__ w_reg, const float* __restrict__ b_reg,
    const float* __restrict__ w_obj, const float* __restrict__ b_obj,
    const float* __restrict__ anchors,  // (B, HWA, 4)
    float* __restrict__ verts,          // (B, HWA, 8)
    float* __restrict__ score,          // (B, HWA)
    int H, int W)
{
    __shared__ float ldsW[21 * 256];
    const int tid = threadIdx.x;
    for (int u = tid; u < 21 * 256; u += 128)
        ldsW[u] = (u < 18 * 256) ? w_reg[u] : w_obj[u - 18 * 256];
    __syncthreads();

    const int HWl = H * W;
    const int m   = blockIdx.x * 128 + tid;
    const int b   = m / HWl;
    const int hw  = m - b * HWl;

    float acc[21];
#pragma unroll
    for (int j = 0; j < 21; ++j) acc[j] = 0.f;

    const unsigned int* trow = (const unsigned int*)(t + (size_t)m * 256);
    for (int cp = 0; cp < 128; ++cp) {
        const unsigned int tw = trow[cp];
        const float f0 = bf2f((unsigned short)(tw & 0xFFFFu));
        const float f1 = bf2f((unsigned short)(tw >> 16));
        const int c = cp * 2;
#pragma unroll
        for (int j = 0; j < 21; ++j)
            acc[j] += f0 * ldsW[j * 256 + c] + f1 * ldsW[j * 256 + c + 1];
    }

    const size_t HWA = (size_t)HWl * AANCH;
#pragma unroll
    for (int a = 0; a < AANCH; ++a) {
        const float dx = acc[a * 6 + 0] + b_reg[a * 6 + 0];
        const float dy = acc[a * 6 + 1] + b_reg[a * 6 + 1];
        float dw = acc[a * 6 + 2] + b_reg[a * 6 + 2];
        float dh = acc[a * 6 + 3] + b_reg[a * 6 + 3];
        const float da = acc[a * 6 + 4] + b_reg[a * 6 + 4];
        const float db = acc[a * 6 + 5] + b_reg[a * 6 + 5];
        const float sc = acc[18 + a] + b_obj[a];

        const size_t p = (size_t)hw * AANCH + a;
        const float* anc = anchors + (b * HWA + p) * 4;
        const float cx = anc[0], cy = anc[1], wa = anc[2], ha = anc[3];

        dw = fminf(fmaxf(dw, -4.f), 4.f);
        dh = fminf(fmaxf(dh, -4.f), 4.f);
        const float w  = wa * expf(dw);
        const float h  = ha * expf(dh);
        const float xx = cx + dx * wa;
        const float yy = cy + dy * ha;
        const float ga = da * w;
        const float gb = db * h;

        float* vo = verts + (b * HWA + p) * 8;
        vo[0] = xx + ga;       vo[1] = yy - h * 0.5f;
        vo[2] = xx + w * 0.5f; vo[3] = yy + gb;
        vo[4] = xx - ga;       vo[5] = yy + h * 0.5f;
        vo[6] = xx - w * 0.5f; vo[7] = yy - gb;
        score[b * HWA + p] = sc;
    }
}

// ---------------------------------------------------------------------------
// Kernel 4: top-K with exact sorted order per (b, level).
// 11-bit radix histogram -> threshold bin -> compact -> bitonic sort (2048).
// ---------------------------------------------------------------------------
__global__ __launch_bounds__(1024) void orpn_topk(
    const float* __restrict__ scoreL,   // (B, N)
    unsigned int* __restrict__ tIdx,    // (10, 2048)
    float* __restrict__ tScore,         // (10, 2048)
    int N, int Kwant, int lvl)
{
    __shared__ unsigned int hist[2048];
    __shared__ unsigned int skey[2048];
    __shared__ unsigned int sidx[2048];
    __shared__ unsigned int cntA, cntB;
    __shared__ int binStar, aboveCnt, needB;

    const int tid = threadIdx.x;
    const int b   = blockIdx.x;
    const int bl  = lvl * 2 + b;
    const float* s = scoreL + (size_t)b * N;

    for (int u = tid; u < 2048; u += 1024) hist[u] = 0u;
    __syncthreads();

    for (int i = tid; i < N; i += 1024)
        atomicAdd(&hist[score_key(s[i]) >> 21], 1u);
    __syncthreads();

    if (tid == 0) {
        int cum = 0;
        binStar = 0; aboveCnt = 0; needB = Kwant;
        for (int bin = 2047; bin >= 0; --bin) {
            const int c = (int)hist[bin];
            if (cum + c >= Kwant) { binStar = bin; aboveCnt = cum; needB = Kwant - cum; break; }
            cum += c;
        }
        cntA = 0u; cntB = 0u;
    }
    __syncthreads();

    for (int i = tid; i < N; i += 1024) {
        const unsigned int key = score_key(s[i]);
        const int bin = (int)(key >> 21);
        if (bin > binStar) {
            const unsigned int pos = atomicAdd(&cntA, 1u);
            skey[pos] = key; sidx[pos] = (unsigned int)i;
        } else if (bin == binStar) {
            const unsigned int p2 = atomicAdd(&cntB, 1u);
            if ((int)p2 < needB) {
                const unsigned int pos = (unsigned int)aboveCnt + p2;
                skey[pos] = key; sidx[pos] = (unsigned int)i;
            }
        }
    }
    __syncthreads();

    for (int u = Kwant + tid; u < 2048; u += 1024) { skey[u] = 0u; sidx[u] = 0u; }
    __syncthreads();

    // bitonic sort, descending by key
    for (int ksz = 2; ksz <= 2048; ksz <<= 1) {
        for (int j = ksz >> 1; j > 0; j >>= 1) {
            for (int i = tid; i < 2048; i += 1024) {
                const int ixj = i ^ j;
                if (ixj > i) {
                    const unsigned int a = skey[i], bb = skey[ixj];
                    const bool up = ((i & ksz) == 0);
                    if (up ? (a < bb) : (a > bb)) {
                        skey[i] = bb; skey[ixj] = a;
                        const unsigned int ti = sidx[i];
                        sidx[i] = sidx[ixj]; sidx[ixj] = ti;
                    }
                }
            }
            __syncthreads();
        }
    }

    for (int k = tid; k < Kwant; k += 1024) {
        const unsigned int id = sidx[k];
        tIdx[(size_t)bl * 2048 + k]   = id;
        tScore[(size_t)bl * 2048 + k] = s[id];
    }
}

// ---------------------------------------------------------------------------
// Kernel 5: gather sorted verts + compute horizontal bounding boxes
// ---------------------------------------------------------------------------
__global__ void orpn_gather_hbb(
    const float* __restrict__ vertsL,      // (B, N, 8)
    const unsigned int* __restrict__ tIdx, // (10, 2048)
    float* __restrict__ tv,                // (10, 2048, 8)
    float* __restrict__ hbb,               // (10, 2048, 4)
    int N, int Kwant, int lvl)
{
    const int k = blockIdx.x * 256 + threadIdx.x;
    if (k >= Kwant) return;
    const int b  = blockIdx.y;
    const int bl = lvl * 2 + b;
    const unsigned int idx = tIdx[(size_t)bl * 2048 + k];
    const float* v = vertsL + ((size_t)b * N + idx) * 8;
    float vv[8];
#pragma unroll
    for (int j = 0; j < 8; ++j) { vv[j] = v[j]; tv[((size_t)bl * 2048 + k) * 8 + j] = vv[j]; }
    const float xmn = fminf(fminf(vv[0], vv[2]), fminf(vv[4], vv[6]));
    const float xmx = fmaxf(fmaxf(vv[0], vv[2]), fmaxf(vv[4], vv[6]));
    const float ymn = fminf(fminf(vv[1], vv[3]), fminf(vv[5], vv[7]));
    const float ymx = fmaxf(fmaxf(vv[1], vv[3]), fmaxf(vv[5], vv[7]));
    float* o = hbb + ((size_t)bl * 2048 + k) * 4;
    o[0] = xmn; o[1] = ymn; o[2] = xmx; o[3] = ymx;
}

// ---------------------------------------------------------------------------
// Kernel 6: NMS suppression bitmask (32x32 box-pair blocks)
// ---------------------------------------------------------------------------
__global__ __launch_bounds__(32) void orpn_nms_mask(
    const float* __restrict__ hbb, unsigned int* __restrict__ mask,
    int Kwant, int lvl)
{
    const int tid = threadIdx.x;
    const int bl  = lvl * 2 + blockIdx.z;
    const int i   = blockIdx.x * 32 + tid;
    const int jb  = blockIdx.y;

    __shared__ float jbox[32][4];
    const int jj = jb * 32 + tid;
    if (jj < Kwant) {
        const float* p = hbb + ((size_t)bl * 2048 + jj) * 4;
        jbox[tid][0] = p[0]; jbox[tid][1] = p[1]; jbox[tid][2] = p[2]; jbox[tid][3] = p[3];
    } else {
        jbox[tid][0] = 0.f; jbox[tid][1] = 0.f; jbox[tid][2] = 0.f; jbox[tid][3] = 0.f;
    }
    __syncthreads();

    if (i >= Kwant) return;
    const float* p = hbb + ((size_t)bl * 2048 + i) * 4;
    const float ix0 = p[0], iy0 = p[1], ix1 = p[2], iy1 = p[3];
    const float areai = (ix1 - ix0) * (iy1 - iy0);

    unsigned int bits = 0u;
#pragma unroll 4
    for (int jl = 0; jl < 32; ++jl) {
        const int j = jb * 32 + jl;
        if (j < Kwant && j > i) {
            const float jx0 = jbox[jl][0], jy0 = jbox[jl][1];
            const float jx1 = jbox[jl][2], jy1 = jbox[jl][3];
            const float areaj = (jx1 - jx0) * (jy1 - jy0);
            const float w = fmaxf(fminf(ix1, jx1) - fmaxf(ix0, jx0), 0.f);
            const float h = fmaxf(fminf(iy1, jy1) - fmaxf(iy0, jy0), 0.f);
            const float inter = w * h;
            const float iou = inter / fmaxf(areai + areaj - inter, 1e-6f);
            if (iou > 0.8f) bits |= (1u << jl);
        }
    }
    mask[((size_t)bl * 2048 + i) * 64 + jb] = bits;
}

// ---------------------------------------------------------------------------
// Kernel 7: sequential NMS scan (reproduces fori_loop suppression order)
// ---------------------------------------------------------------------------
__global__ __launch_bounds__(64) void orpn_nms_scan(
    const unsigned int* __restrict__ mask, unsigned int* __restrict__ keep,
    int Kwant, int lvl)
{
    __shared__ unsigned int remv[64];
    const int tid = threadIdx.x;
    const int bl  = lvl * 2 + blockIdx.x;
    const int nb  = (Kwant + 31) / 32;
    remv[tid] = 0u;
    __syncthreads();

    for (int i = 0; i < Kwant; ++i) {
        const unsigned int sup = (remv[i >> 5] >> (i & 31)) & 1u;
        __syncthreads();
        if (!sup && tid < nb)
            remv[tid] |= mask[((size_t)bl * 2048 + i) * 64 + tid];
        if (tid == 0)
            keep[(size_t)bl * 2048 + i] = sup ? 0u : 1u;
        __syncthreads();
    }
}

// ---------------------------------------------------------------------------
// Kernel 8: final writeout (B, 8768, 9), keep-masked, levels concatenated
// ---------------------------------------------------------------------------
__global__ void orpn_writeout(
    const float* __restrict__ tv, const float* __restrict__ tScore,
    const unsigned int* __restrict__ keep, float* __restrict__ out,
    int Kwant, int rowStart, int lvl)
{
    const int k = blockIdx.x * 256 + threadIdx.x;
    if (k >= Kwant) return;
    const int b  = blockIdx.y;
    const int bl = lvl * 2 + b;
    const float kf = keep[(size_t)bl * 2048 + k] ? 1.f : 0.f;
    float* o = out + ((size_t)b * TOTROWS + rowStart + k) * 9;
#pragma unroll
    for (int j = 0; j < 8; ++j)
        o[j] = tv[((size_t)bl * 2048 + k) * 8 + j] * kf;
    o[8] = tScore[(size_t)bl * 2048 + k] * kf;
}

// ---------------------------------------------------------------------------
// Host launcher
// ---------------------------------------------------------------------------
extern "C" void kernel_launch(void* const* d_in, const int* in_sizes, int n_in,
                              void* d_out, int out_size, void* d_ws, size_t ws_size,
                              hipStream_t stream) {
    (void)in_sizes; (void)n_in; (void)out_size; (void)ws_size;

    static const int Hs[5] = {256, 128, 64, 32, 16};
    const int B = 2;
    int HW[5], Kw[5], rowStart[5];
    int rs = 0;
    for (int l = 0; l < 5; ++l) {
        HW[l] = Hs[l] * Hs[l];
        const int hwa = HW[l] * AANCH;
        Kw[l] = hwa < 2000 ? hwa : 2000;
        rowStart[l] = rs;
        rs += Kw[l];
    }

    // workspace layout (256B aligned)
    size_t off = 0;
    auto alloc = [&](size_t bytes) { size_t o = off; off = (off + bytes + 255) & ~(size_t)255; return o; };
    const size_t o_wB = alloc((size_t)5 * 256 * KDIM * 2);
    size_t o_t[5], o_verts[5], o_score[5];
    for (int l = 0; l < 5; ++l) o_t[l]     = alloc((size_t)B * HW[l] * 256 * 2);
    for (int l = 0; l < 5; ++l) o_verts[l] = alloc((size_t)B * HW[l] * AANCH * 8 * 4);
    for (int l = 0; l < 5; ++l) o_score[l] = alloc((size_t)B * HW[l] * AANCH * 4);
    const size_t o_tidx = alloc((size_t)10 * 2048 * 4);
    const size_t o_tsc  = alloc((size_t)10 * 2048 * 4);
    const size_t o_tv   = alloc((size_t)10 * 2048 * 8 * 4);
    const size_t o_hbb  = alloc((size_t)10 * 2048 * 4 * 4);
    const size_t o_mask = alloc((size_t)10 * 2048 * 64 * 4);
    const size_t o_keep = alloc((size_t)10 * 2048 * 4);

    char* ws = (char*)d_ws;
    unsigned short* wB    = (unsigned short*)(ws + o_wB);
    unsigned int*   tIdx  = (unsigned int*)(ws + o_tidx);
    float*          tSc   = (float*)(ws + o_tsc);
    float*          tv    = (float*)(ws + o_tv);
    float*          hbb   = (float*)(ws + o_hbb);
    unsigned int*   mask  = (unsigned int*)(ws + o_mask);
    unsigned int*   keep  = (unsigned int*)(ws + o_keep);
    float*          out   = (float*)d_out;

    const float* w_conv = (const float*)d_in[10];
    const float* b_conv = (const float*)d_in[11];
    const float* w_reg  = (const float*)d_in[12];
    const float* b_reg  = (const float*)d_in[13];
    const float* w_obj  = (const float*)d_in[14];
    const float* b_obj  = (const float*)d_in[15];

    // 1) repack weights to bf16 (tap-major K)
    {
        const int total = 5 * 256 * KDIM;
        orpn_repack_w<<<(total + 255) / 256, 256, 0, stream>>>(w_conv, wB);
    }

    for (int l = 0; l < 5; ++l) {
        const float* x   = (const float*)d_in[2 * l];
        const float* anc = (const float*)d_in[2 * l + 1];
        unsigned short* tL = (unsigned short*)(ws + o_t[l]);
        float* vertsL = (float*)(ws + o_verts[l]);
        float* scoreL = (float*)(ws + o_score[l]);
        const int H = Hs[l], W = Hs[l];
        const int Mtot = B * HW[l];

        // 2) 3x3 conv + ReLU (WMMA implicit GEMM, 128x128 tile)
        orpn_conv3x3_wmma<<<dim3(Mtot / 128, 2), 256, 0, stream>>>(
            x, wB + (size_t)l * 256 * KDIM, b_conv + l * 256, tL, H, W);

        // 3) 1x1 heads + decode
        orpn_head_decode<<<Mtot / 128, 128, 0, stream>>>(
            tL, w_reg + (size_t)l * 18 * 256, b_reg + l * 18,
            w_obj + (size_t)l * 3 * 256, b_obj + l * 3,
            anc, vertsL, scoreL, H, W);

        // 4) top-K (sorted)
        orpn_topk<<<B, 1024, 0, stream>>>(scoreL, tIdx, tSc, HW[l] * AANCH, Kw[l], l);

        // 5) gather + HBB
        orpn_gather_hbb<<<dim3((Kw[l] + 255) / 256, B), 256, 0, stream>>>(
            vertsL, tIdx, tv, hbb, HW[l] * AANCH, Kw[l], l);

        // 6) NMS mask
        const int nb = (Kw[l] + 31) / 32;
        orpn_nms_mask<<<dim3(nb, nb, B), 32, 0, stream>>>(hbb, mask, Kw[l], l);

        // 7) NMS sequential scan
        orpn_nms_scan<<<B, 64, 0, stream>>>(mask, keep, Kw[l], l);

        // 8) writeout
        orpn_writeout<<<dim3((Kw[l] + 255) / 256, B), 256, 0, stream>>>(
            tv, tSc, keep, out, Kw[l], rowStart[l], l);
    }
}
